// LongRangeInteraction_89326729822708
// MI455X (gfx1250) — matrix operand order repaired and checked by reading, not compiled, not run on hardware
//
#include <hip/hip_runtime.h>
#include <hip/hip_bf16.h>

// MI455X / gfx1250 implementation.
// Pipeline: segptr -> pack(f16 transposes) -> phase tables -> MLP L1 (VALU)
//        -> MLP L2+L3 (WMMA) -> segment-sum s=E^H h (WMMA) -> g=s*filt pack
//        -> out = E (g) (WMMA, complex), all on `stream`, scratch from d_ws.

typedef __attribute__((ext_vector_type(16))) _Float16 v16h;
typedef __attribute__((ext_vector_type(8)))  _Float16 v8h;
typedef __attribute__((ext_vector_type(8)))  float    v8f;

#define D_PET   128
#define NB      2048
#define NK      256
#define NSEG    16
#define NT_ROWS (NB + 16)   // atom-major tables padded so 16-row tiles may overhang

// ---------------- WMMA fragment helpers (wave32, 16x16x32 f16) ----------------

__device__ __forceinline__ v8f wmma_f16(v16h a, v16h b, v8f c) {
  return __builtin_amdgcn_wmma_f32_16x16x32_f16(false, a, false, b, (short)0, c,
                                                false, false);
}

// A fragment (16x32, f16) from row-major [M][K] memory, tile at (row0, k0).
// lane l<16: row l, K = k0+{0..7, 16..23}; lane l>=16: row l-16, K = k0+{8..15, 24..31}
__device__ __forceinline__ v16h load_a_frag(const _Float16* __restrict__ base,
                                            int stride, int row0, int k0) {
  int l = threadIdx.x & 31;
  const _Float16* p = base + (size_t)(row0 + (l & 15)) * stride + k0 + ((l & 16) ? 8 : 0);
  union { v16h v; v8h h[2]; } u;
  u.h[0] = *(const v8h*)(p);
  u.h[1] = *(const v8h*)(p + 16);
  return u.v;
}

// Masked A fragment: K-index (absolute column) restricted to [colLo, colHi), else 0.
__device__ __forceinline__ v16h load_a_frag_masked(const _Float16* __restrict__ base,
                                                   int stride, int row0, int k0,
                                                   int colLo, int colHi) {
  int l = threadIdx.x & 31;
  const _Float16* p = base + (size_t)(row0 + (l & 15)) * stride;
  int cbase = k0 + ((l & 16) ? 8 : 0);
  v16h v;
#pragma unroll
  for (int j = 0; j < 16; ++j) {
    int c = cbase + ((j < 8) ? j : (8 + j));   // j>=8 -> +16+(j-8)
    v[j] = (c >= colLo && c < colHi) ? p[c] : (_Float16)0.0f;
  }
  return v;
}

// B fragment (32x16, f16) from transposed [N][K] memory, tile at (n0, k0).
// lane l<16: N = n0+l, K = k0+{0..7, 16..23}; lane l>=16: N = n0+l-16, K = k0+{8..15, 24..31}
__device__ __forceinline__ v16h load_b_frag(const _Float16* __restrict__ base,
                                            int stride, int n0, int k0) {
  int l = threadIdx.x & 31;
  const _Float16* p = base + (size_t)(n0 + (l & 15)) * stride + k0 + ((l & 16) ? 8 : 0);
  union { v16h v; v8h h[2]; } u;
  u.h[0] = *(const v8h*)(p);
  u.h[1] = *(const v8h*)(p + 16);
  return u.v;
}

__device__ __forceinline__ v16h load_b_frag_masked(const _Float16* __restrict__ base,
                                                   int stride, int n0, int k0,
                                                   int colLo, int colHi) {
  int l = threadIdx.x & 31;
  const _Float16* p = base + (size_t)(n0 + (l & 15)) * stride;
  int cbase = k0 + ((l & 16) ? 8 : 0);
  v16h v;
#pragma unroll
  for (int j = 0; j < 16; ++j) {
    int c = cbase + ((j < 8) ? j : (8 + j));
    v[j] = (c >= colLo && c < colHi) ? p[c] : (_Float16)0.0f;
  }
  return v;
}

// C/D (16x16 f32) store to row-major [stride] f32.
__device__ __forceinline__ void store_c(float* __restrict__ base, int stride,
                                        int row0, int col0, v8f acc) {
  int l = threadIdx.x & 31;
  int col = col0 + (l & 15);
  int rbase = row0 + ((l & 16) ? 8 : 0);
#pragma unroll
  for (int r = 0; r < 8; ++r) base[(size_t)(rbase + r) * stride + col] = acc[r];
}

__device__ __forceinline__ float silu(float v) { return v / (1.0f + __expf(-v)); }

// ---------------- Kernel 0: segment boundaries (batch is sorted) ----------------

__global__ void k_segptr(const int* __restrict__ batch, int* __restrict__ segPtr) {
  int t = threadIdx.x;
  if (t > NSEG) return;
  int lo = 0, hi = NB;                       // first i with batch[i] >= t
  while (lo < hi) { int mid = (lo + hi) >> 1; if (batch[mid] < t) lo = mid + 1; else hi = mid; }
  segPtr[t] = lo;
}

// ---------------- Kernel 1: f16 packs (W2^T, W3^T, h^T) ----------------

__global__ void k_pack(const float* __restrict__ W2, const float* __restrict__ W3,
                       const float* __restrict__ h,
                       _Float16* __restrict__ W2T, _Float16* __restrict__ W3T,
                       _Float16* __restrict__ hT) {
  int idx = blockIdx.x * blockDim.x + threadIdx.x;
  if (idx < D_PET * D_PET) {                 // weights: [e][d] = W[d][e]
    int e = idx / D_PET, d = idx % D_PET;
    W2T[idx] = (_Float16)W2[d * D_PET + e];
    W3T[idx] = (_Float16)W3[d * D_PET + e];
  }
  int tot = D_PET * NB;                      // hT: [d][n]
  for (int i = idx; i < tot; i += gridDim.x * blockDim.x) {
    int d = i / NB, n = i % NB;
    hT[i] = (_Float16)h[(size_t)n * D_PET + d];
  }
}

// ---------------- Kernel 2: phase tables (both orientations, f16) ----------------

__global__ void k_phase(const float* __restrict__ kvec, const float* __restrict__ pos,
                        const int* __restrict__ batch,
                        _Float16* __restrict__ cosNT, _Float16* __restrict__ sinNT,
                        _Float16* __restrict__ msinNT,
                        _Float16* __restrict__ cosKT, _Float16* __restrict__ msinKT) {
  int idx = blockIdx.x * blockDim.x + threadIdx.x;
  if (idx >= NT_ROWS * NK) return;
  int n = idx / NK, k = idx % NK;
  if (n >= NB) {                             // zero the 16 pad rows
    cosNT[idx] = (_Float16)0.0f; sinNT[idx] = (_Float16)0.0f; msinNT[idx] = (_Float16)0.0f;
    return;
  }
  int b = batch[n];
  const float* kv = kvec + ((size_t)b * NK + k) * 3;
  const float* p  = pos + (size_t)n * 3;
  float ph = kv[0] * p[0] + kv[1] * p[1] + kv[2] * p[2];
  float c = __cosf(ph), s = __sinf(ph);
  cosNT[idx]  = (_Float16)c;
  sinNT[idx]  = (_Float16)s;
  msinNT[idx] = (_Float16)(-s);
  size_t kt = (size_t)k * NB + n;            // k-major, atom-contiguous
  cosKT[kt]  = (_Float16)c;
  msinKT[kt] = (_Float16)(-s);
}

// ---------------- Kernel 3: MLP layer 1 (K=3, plain VALU) ----------------

__global__ void k_mlp1(const float* __restrict__ kvec, const float* __restrict__ W1,
                       const float* __restrict__ b1, _Float16* __restrict__ x1) {
  int idx = blockIdx.x * blockDim.x + threadIdx.x;      // over 4096*128
  if (idx >= NSEG * NK * D_PET) return;
  int d = idx % D_PET, r = idx / D_PET;
  const float* kv = kvec + (size_t)r * 3;
  float v = kv[0] * W1[0 * D_PET + d] + kv[1] * W1[1 * D_PET + d] +
            kv[2] * W1[2 * D_PET + d] + b1[d];
  x1[idx] = (_Float16)silu(v);
}

// ---------------- Kernel 4: MLP layers 2+3 fused (WMMA, LDS staging) ----------------

__global__ void k_mlp23(const _Float16* __restrict__ x1, const _Float16* __restrict__ W2T,
                        const float* __restrict__ b2, const _Float16* __restrict__ W3T,
                        const float* __restrict__ b3, float* __restrict__ filt) {
  __shared__ _Float16 x2[16 * D_PET];
  int row0 = blockIdx.x * 16;                // 256 row tiles of (b,k) rows
  int l = threadIdx.x & 31;
  v16h a[4];
#pragma unroll
  for (int kc = 0; kc < 4; ++kc) a[kc] = load_a_frag(x1, D_PET, row0, kc * 32);
#pragma unroll
  for (int dt = 0; dt < 8; ++dt) {           // layer 2: x2 = silu(x1*W2 + b2)
    v8f acc = 0;
#pragma unroll
    for (int kc = 0; kc < 4; ++kc)
      acc = wmma_f16(a[kc], load_b_frag(W2T, D_PET, dt * 16, kc * 32), acc);
    int col = dt * 16 + (l & 15);
    int rbase = (l & 16) ? 8 : 0;
    float bias = b2[col];
#pragma unroll
    for (int r = 0; r < 8; ++r)
      x2[(rbase + r) * D_PET + col] = (_Float16)silu(acc[r] + bias);
  }
  __syncthreads();
#pragma unroll
  for (int kc = 0; kc < 4; ++kc) a[kc] = load_a_frag(x2, D_PET, 0, kc * 32);
#pragma unroll
  for (int dt = 0; dt < 8; ++dt) {           // layer 3: filt = x2*W3 + b3
    v8f acc = 0;
#pragma unroll
    for (int kc = 0; kc < 4; ++kc)
      acc = wmma_f16(a[kc], load_b_frag(W3T, D_PET, dt * 16, kc * 32), acc);
    int col = dt * 16 + (l & 15);
    int rbase = (l & 16) ? 8 : 0;
    float bias = b3[col];
#pragma unroll
    for (int r = 0; r < 8; ++r)
      filt[(size_t)(row0 + rbase + r) * D_PET + col] = acc[r] + bias;
  }
}

// ---------------- Kernel 5: s[b,k,d] = sum_{n in seg b} (cos - i sin)*h (WMMA) ----------------

__global__ void k_segsum(const _Float16* __restrict__ cosKT, const _Float16* __restrict__ msinKT,
                         const _Float16* __restrict__ hT, const int* __restrict__ segPtr,
                         float* __restrict__ sRe, float* __restrict__ sIm) {
  int k0 = blockIdx.x * 16;                  // 16 k-tiles
  int d0 = blockIdx.y * 64;                  // 2 d-halves, 4 tiles each
  int b  = blockIdx.z;
  int segStart = segPtr[b], segEnd = segPtr[b + 1];
  v8f accRe[4], accIm[4];
#pragma unroll
  for (int dt = 0; dt < 4; ++dt) { accRe[dt] = 0; accIm[dt] = 0; }

  for (int n0 = segStart & ~31; n0 < segEnd; n0 += 32) {   // 32-aligned atom chunks
    bool full = (n0 >= segStart) && (n0 + 32 <= segEnd);   // uniform branch
    v16h aC, aS;
    if (full) {
      aC = load_a_frag(cosKT,  NB, k0, n0);
      aS = load_a_frag(msinKT, NB, k0, n0);
    } else {
      aC = load_a_frag_masked(cosKT,  NB, k0, n0, segStart, segEnd);
      aS = load_a_frag_masked(msinKT, NB, k0, n0, segStart, segEnd);
    }
#pragma unroll
    for (int dt = 0; dt < 4; ++dt) {
      v16h bh = full ? load_b_frag(hT, NB, d0 + dt * 16, n0)
                     : load_b_frag_masked(hT, NB, d0 + dt * 16, n0, segStart, segEnd);
      accRe[dt] = wmma_f16(aC, bh, accRe[dt]);
      accIm[dt] = wmma_f16(aS, bh, accIm[dt]);   // A = -sin  =>  Im(s) directly
    }
  }
  float* sReB = sRe + (size_t)b * NK * D_PET;
  float* sImB = sIm + (size_t)b * NK * D_PET;
#pragma unroll
  for (int dt = 0; dt < 4; ++dt) {
    store_c(sReB, D_PET, k0, d0 + dt * 16, accRe[dt]);
    store_c(sImB, D_PET, k0, d0 + dt * 16, accIm[dt]);
  }
}

// ---------------- Kernel 6: g = s * filt, packed f16 transposed [b][d][k] ----------------

__global__ void k_gpack(const float* __restrict__ sRe, const float* __restrict__ sIm,
                        const float* __restrict__ filt,
                        _Float16* __restrict__ gRe, _Float16* __restrict__ gIm) {
  int idx = blockIdx.x * blockDim.x + threadIdx.x;     // over [b][d][k]
  if (idx >= NSEG * D_PET * NK) return;
  int k = idx % NK, d = (idx / NK) % D_PET, b = idx / (NK * D_PET);
  size_t src = ((size_t)b * NK + k) * D_PET + d;
  float f = filt[src];
  gRe[idx] = (_Float16)(sRe[src] * f);
  gIm[idx] = (_Float16)(sIm[src] * f);
}

// ---------------- Kernel 7: out[n,d] = sum_k e^{i phi} * g  (complex, WMMA) ----------------

__global__ void k_out(const _Float16* __restrict__ cosNT, const _Float16* __restrict__ sinNT,
                      const _Float16* __restrict__ msinNT,
                      const _Float16* __restrict__ gRe, const _Float16* __restrict__ gIm,
                      const int* __restrict__ segPtr, float* __restrict__ out) {
  int b = blockIdx.z;
  int segStart = segPtr[b], segEnd = segPtr[b + 1];
  int row0 = segStart + blockIdx.x * 16;     // 16-atom tile inside segment b
  if (row0 >= segEnd) return;                // uniform exit
  int d0 = blockIdx.y * 16;
  const _Float16* gReB = gRe + (size_t)b * D_PET * NK;
  const _Float16* gImB = gIm + (size_t)b * D_PET * NK;
  v8f accRe = 0, accIm = 0;
#pragma unroll
  for (int kc = 0; kc < 8; ++kc) {           // K = 256 over k
    int k0 = kc * 32;
    v16h aC = load_a_frag(cosNT,  NK, row0, k0);   // overhang rows: padded, discarded
    v16h aS = load_a_frag(sinNT,  NK, row0, k0);
    v16h aM = load_a_frag(msinNT, NK, row0, k0);
    v16h bR = load_b_frag(gReB, NK, d0, k0);
    v16h bI = load_b_frag(gImB, NK, d0, k0);
    accRe = wmma_f16(aC, bR, accRe);
    accRe = wmma_f16(aM, bI, accRe);         // re = cos*gRe - sin*gIm
    accIm = wmma_f16(aS, bR, accIm);
    accIm = wmma_f16(aC, bI, accIm);         // im = sin*gRe + cos*gIm
  }
  int l = threadIdx.x & 31;
  int col = d0 + (l & 15);
  int rbase = row0 + ((l & 16) ? 8 : 0);
#pragma unroll
  for (int r = 0; r < 8; ++r) {
    int row = rbase + r;
    if (row < segEnd) {                      // mask tile overhang (store-only divergence)
      size_t o = ((size_t)row * D_PET + col) * 2;   // complex64 interleaved
      float2 v; v.x = accRe[r]; v.y = accIm[r];
      *(float2*)(out + o) = v;
    }
  }
}

// ---------------- Host side ----------------

extern "C" void kernel_launch(void* const* d_in, const int* in_sizes, int n_in,
                              void* d_out, int out_size, void* d_ws, size_t ws_size,
                              hipStream_t stream) {
  const float* kvec  = (const float*)d_in[0];
  const float* pos   = (const float*)d_in[1];
  const float* h     = (const float*)d_in[2];
  const float* W1    = (const float*)d_in[3];
  const float* b1    = (const float*)d_in[4];
  const float* W2    = (const float*)d_in[5];
  const float* b2    = (const float*)d_in[6];
  const float* W3    = (const float*)d_in[7];
  const float* b3    = (const float*)d_in[8];
  const int*   batch = (const int*)d_in[9];
  float* out = (float*)d_out;

  char* ws = (char*)d_ws;
  auto carve = [&](size_t bytes) -> char* {
    char* p = ws; ws += (bytes + 255) & ~(size_t)255; return p;
  };
  int*      segPtr = (int*)      carve(32 * sizeof(int));
  _Float16* cosNT  = (_Float16*) carve((size_t)NT_ROWS * NK * 2);
  _Float16* sinNT  = (_Float16*) carve((size_t)NT_ROWS * NK * 2);
  _Float16* msinNT = (_Float16*) carve((size_t)NT_ROWS * NK * 2);
  _Float16* cosKT  = (_Float16*) carve((size_t)NK * NB * 2);
  _Float16* msinKT = (_Float16*) carve((size_t)NK * NB * 2);
  _Float16* hT     = (_Float16*) carve((size_t)D_PET * NB * 2);
  _Float16* W2T    = (_Float16*) carve((size_t)D_PET * D_PET * 2);
  _Float16* W3T    = (_Float16*) carve((size_t)D_PET * D_PET * 2);
  _Float16* x1     = (_Float16*) carve((size_t)NSEG * NK * D_PET * 2);
  float*    filt   = (float*)    carve((size_t)NSEG * NK * D_PET * 4);
  float*    sRe    = (float*)    carve((size_t)NSEG * NK * D_PET * 4);
  float*    sIm    = (float*)    carve((size_t)NSEG * NK * D_PET * 4);
  _Float16* gRe    = (_Float16*) carve((size_t)NSEG * D_PET * NK * 2);
  _Float16* gIm    = (_Float16*) carve((size_t)NSEG * D_PET * NK * 2);
  (void)ws_size; (void)in_sizes; (void)n_in; (void)out_size;

  k_segptr<<<1, 32, 0, stream>>>(batch, segPtr);
  k_pack<<<1024, 256, 0, stream>>>(W2, W3, h, W2T, W3T, hT);
  {
    int tot = NT_ROWS * NK;
    k_phase<<<(tot + 255) / 256, 256, 0, stream>>>(kvec, pos, batch,
                                                   cosNT, sinNT, msinNT, cosKT, msinKT);
  }
  {
    int tot = NSEG * NK * D_PET;
    k_mlp1<<<(tot + 255) / 256, 256, 0, stream>>>(kvec, W1, b1, x1);
  }
  k_mlp23<<<NSEG * NK / 16, 32, 0, stream>>>(x1, W2T, b2, W3T, b3, filt);
  k_segsum<<<dim3(NK / 16, 2, NSEG), 32, 0, stream>>>(cosKT, msinKT, hT, segPtr, sRe, sIm);
  {
    int tot = NSEG * D_PET * NK;
    k_gpack<<<(tot + 255) / 256, 256, 0, stream>>>(sRe, sIm, filt, gRe, gIm);
  }
  k_out<<<dim3(NB / 16, D_PET / 16, NSEG), 32, 0, stream>>>(cosNT, sinNT, msinNT,
                                                            gRe, gIm, segPtr, out);
}
